// PairedKidneyModel_91216515432550
// MI455X (gfx1250) — compile-verified, matrix operand order checked by reading, not run on hardware
//
#include <hip/hip_runtime.h>

#define HD 64
#define NLAYERS 3
#define NEG_SLOPE 0.2f
#define LN_EPS 1e-5f

typedef __attribute__((ext_vector_type(16))) _Float16 v16h;
typedef __attribute__((ext_vector_type(8)))  float    v8f;
typedef __attribute__((ext_vector_type(4)))  unsigned v4u;
typedef __attribute__((ext_vector_type(4)))  float    v4f;

// ---- monotone float<->uint key for atomic max over signed floats ----
__device__ __forceinline__ unsigned fkey(float f) {
  unsigned u = __float_as_uint(f);
  return (u & 0x80000000u) ? ~u : (u | 0x80000000u);
}
__device__ __forceinline__ float funkey(unsigned k) {
  unsigned u = (k & 0x80000000u) ? (k ^ 0x80000000u) : ~k;
  return __uint_as_float(u);
}

// ---- fuse the two embedding linears: Wf[2][64] = W1@W2, bf = b1@W2 + b2 ----
__global__ void gat_fuse_emb(const float* __restrict__ W1, const float* __restrict__ b1,
                             const float* __restrict__ W2, const float* __restrict__ b2,
                             float* __restrict__ Wf, float* __restrict__ bf) {
  int j = threadIdx.x;
  if (j >= HD) return;
  float w0 = 0.f, w1 = 0.f, bb = 0.f;
  for (int k = 0; k < HD; ++k) {
    float w2 = W2[k * HD + j];
    w0 += W1[k] * w2;          // W1[0][k]
    w1 += W1[HD + k] * w2;     // W1[1][k]
    bb += b1[k] * w2;
  }
  Wf[j] = w0;
  Wf[HD + j] = w1;
  bf[j] = bb + b2[j];
}

// ---- per active node: build 2-feature and apply fused embedding -> x0[A][64] ----
__global__ void gat_embed(const float* __restrict__ arr, const float* __restrict__ dep,
                          const float* __restrict__ hard, const int* __restrict__ mi,
                          const int* __restrict__ tstep, const float* __restrict__ Wf,
                          const float* __restrict__ bf, float* __restrict__ x0, int nA) {
  int i = blockIdx.x * blockDim.x + threadIdx.x;
  if (i >= nA) return;
  int g = mi[i];
  float t = (float)tstep[0];
  float a = arr[g];
  float p = (t - a) / (dep[g] - a);
  float hh = hard[g];
  float* row = x0 + (size_t)i * HD;
  for (int j = 0; j < HD; ++j)
    row[j] = p * Wf[j] + hh * Wf[HD + j] + bf[j];
}

// ---- fp32 -> fp16 conversion (4 elems/thread) ----
__global__ void gat_f2h(const float* __restrict__ in, _Float16* __restrict__ out, int n) {
  int i = (blockIdx.x * blockDim.x + threadIdx.x) * 4;
  if (i >= n) return;
  union { _Float16 h[4]; unsigned long long u; } o;
  o.h[0] = (_Float16)in[i + 0];
  o.h[1] = (_Float16)in[i + 1];
  o.h[2] = (_Float16)in[i + 2];
  o.h[3] = (_Float16)in[i + 3];
  *(unsigned long long*)(out + i) = o.u;
}

// ---- pack W[64][64] (fp32, row=k) into WMMA-B-fragment-native f16 dwords ----
// dword index = ((ntile*2 + kchunk)*32 + lane)*8 + v
// holds halfs (W[k][n], W[k+1][n]) with n = ntile*16 + lane%16,
// k = kchunk*32 + (v>=4 ? 16:0) + (lane/16)*8 + 2*(v%4)
__global__ void gat_pack_w(const float* __restrict__ W, unsigned* __restrict__ wpack) {
  int idx = blockIdx.x * blockDim.x + threadIdx.x;
  if (idx >= 4 * 2 * 32 * 8) return;
  int v = idx & 7;
  int lane = (idx >> 3) & 31;
  int c = (idx >> 8) & 1;
  int ntile = idx >> 9;
  int n = ntile * 16 + (lane & 15);
  int k = c * 32 + ((v & 4) ? 16 : 0) + ((lane >> 4) << 3) + ((v & 3) << 1);
  union { _Float16 h[2]; unsigned u; } o;
  o.h[0] = (_Float16)W[k * HD + n];
  o.h[1] = (_Float16)W[(k + 1) * HD + n];
  wpack[idx] = o.u;
}

// ---- hw = h @ W via v_wmma_f32_16x16x32_f16; block = 8 waves = 32 rows x 64 cols ----
// A tile staged global->LDS with CDNA5 async copy (ASYNCcnt), fragments via ds_load_b128.
__global__ __launch_bounds__(256)
void gat_gemm_wmma(const _Float16* __restrict__ h16, const unsigned* __restrict__ wpack,
                   float* __restrict__ hw, int nA) {
  __shared__ _Float16 atile[32 * HD];  // 4 KB A tile
  const int tid = threadIdx.x;
  const int rowbase = blockIdx.x * 32;

  // async global->LDS stage: 256 threads x 16B = 4KB tile
  {
    int r = tid >> 3;
    int c = tid & 7;
    int row = rowbase + r;
    if (row >= nA) row = nA - 1;
    unsigned long long gaddr =
        (unsigned long long)(const void*)(h16 + (size_t)row * HD) + (unsigned)(c * 16);
    // low 32 bits of a generic pointer to LDS are the raw LDS byte address
    unsigned lds_addr = (unsigned)(size_t)(void*)atile + (unsigned)((r * 8 + c) * 16);
    asm volatile("global_load_async_to_lds_b128 %0, %1, off"
                 :: "v"(lds_addr), "v"(gaddr) : "memory");
    asm volatile("s_wait_asynccnt 0x0" ::: "memory");
  }
  __syncthreads();

  const int wave = tid >> 5;
  const int lane = tid & 31;
  const int wrow = (wave & 1) << 4;   // row sub-tile 0/16
  const int ntile = wave >> 1;        // output col tile 0..3
  const int m = lane & 15;
  const int khalf = lane >> 4;

  union Frag { v16h h; v4u u[2]; };
  v8f acc = {};
#pragma unroll
  for (int c = 0; c < 2; ++c) {  // K = 64 in two 32-wide chunks
    Frag a, b;
    // A fragment: 16-bit 16x32 layout. Lane holds row m, K-half khalf.
    // u[0] = halfs k..k+7 at k = c*32 + khalf*8 ; u[1] = same +16
    const v4u* ap = (const v4u*)(atile + (wrow + m) * HD + c * 32 + khalf * 8);
    a.u[0] = ap[0];
    a.u[1] = ap[2];
    // B fragment: 8 contiguous dwords from the packed weight buffer
    const v4u* bp = (const v4u*)(wpack + (((ntile * 2 + c) * 32 + lane) << 3));
    b.u[0] = bp[0];
    b.u[1] = bp[1];
    acc = __builtin_amdgcn_wmma_f32_16x16x32_f16(false, a.h, false, b.h,
                                                 (short)0, acc, false, false);
  }

  // C/D layout: VGPR r -> row r (lanes 0-15) / row r+8 (lanes 16-31), col = lane%16
  const int coln = ntile * 16 + m;
  const int rbase = rowbase + wrow + ((lane < 16) ? 0 : 8);
  if (rowbase + 32 <= nA) {   // uniform fast path: unconditional stores
#pragma unroll
    for (int r = 0; r < 8; ++r)
      hw[(size_t)(rbase + r) * HD + coln] = acc[r];
  } else {
#pragma unroll
    for (int r = 0; r < 8; ++r) {
      int row = rbase + r;
      if (row < nA) hw[(size_t)row * HD + coln] = acc[r];
    }
  }
}

// ---- attention projections a_s = hw@att_src, a_d = hw@att_dst (float4 loads) ----
__global__ void gat_attn(const float* __restrict__ hw, const float* __restrict__ as_w,
                         const float* __restrict__ ad_w, float* __restrict__ a_s,
                         float* __restrict__ a_d, int nA) {
  int i = blockIdx.x * blockDim.x + threadIdx.x;
  if (i >= nA) return;
  const v4f* row = (const v4f*)(hw + (size_t)i * HD);
  const v4f* ws = (const v4f*)as_w;
  const v4f* wd = (const v4f*)ad_w;
  float ss = 0.f, sd = 0.f;
#pragma unroll
  for (int j = 0; j < HD / 4; ++j) {
    v4f v = row[j];
    v4f s = ws[j];
    v4f d = wd[j];
    ss += v.x * s.x + v.y * s.y + v.z * s.z + v.w * s.w;
    sd += v.x * d.x + v.y * d.y + v.z * d.z + v.w * d.w;
  }
  a_s[i] = ss;
  a_d[i] = sd;
}

// ---- edge pass 1: leaky-relu logit + segment max (atomic u32 on monotone key) ----
__global__ void gat_edge_max(const int* __restrict__ esrc, const int* __restrict__ edst,
                             int E, int nA, const float* __restrict__ a_s,
                             const float* __restrict__ a_d, float* __restrict__ eraw,
                             unsigned* __restrict__ mseg) {
  int i = blockIdx.x * blockDim.x + threadIdx.x;
  int tot = E + nA;
  if (i >= tot) return;
  int s = (i < E) ? esrc[i] : (i - E);
  int d = (i < E) ? edst[i] : (i - E);
  float e = a_s[s] + a_d[d];
  e = (e > 0.f) ? e : NEG_SLOPE * e;
  eraw[i] = e;
  atomicMax(mseg + d, fkey(e));
}

// ---- edge pass 2: exp(e - max) + segment sum ----
__global__ void gat_edge_exp(const int* __restrict__ edst, int E, int nA,
                             const unsigned* __restrict__ mseg, float* __restrict__ eraw,
                             float* __restrict__ denom) {
  int i = blockIdx.x * blockDim.x + threadIdx.x;
  int tot = E + nA;
  if (i >= tot) return;
  int d = (i < E) ? edst[i] : (i - E);
  float v = __expf(eraw[i] - funkey(mseg[d]));
  eraw[i] = v;
  atomicAdd(denom + d, v);
}

// ---- edge pass 3: hacc[dst] += alpha * hw[src]; one thread per (edge, feature) ----
__global__ void gat_edge_aggr(const int* __restrict__ esrc, const int* __restrict__ edst,
                              int E, int nA, const float* __restrict__ ee,
                              const float* __restrict__ denom, const float* __restrict__ hw,
                              float* __restrict__ hacc) {
  int t = blockIdx.x * blockDim.x + threadIdx.x;
  int tot = (E + nA) * HD;
  if (t >= tot) return;
  int i = t >> 6;
  int j = t & (HD - 1);
  int s = (i < E) ? esrc[i] : (i - E);
  int d = (i < E) ? edst[i] : (i - E);
  float alpha = ee[i] / denom[d];
  atomicAdd(hacc + (size_t)d * HD + j, hw[(size_t)s * HD + j] * alpha);
}

// ---- bias + optional relu -> next-layer h ----
__global__ void gat_bias_act(const float* __restrict__ hacc, const float* __restrict__ bias,
                             float* __restrict__ hout, int n, int do_relu) {
  int t = blockIdx.x * blockDim.x + threadIdx.x;
  if (t >= n) return;
  float v = hacc[t] + bias[t & (HD - 1)];
  if (do_relu) v = fmaxf(v, 0.f);
  hout[t] = v;
}

// ---- residual + layernorm + sigmoid head + scatter ----
__global__ void gat_final(const float* __restrict__ x0, const float* __restrict__ h,
                          const int* __restrict__ mi, const float* __restrict__ selW,
                          const float* __restrict__ selb, float* __restrict__ out, int nA) {
  int i = blockIdx.x * blockDim.x + threadIdx.x;
  if (i >= nA) return;
  const float* px = x0 + (size_t)i * HD;
  const float* ph = h + (size_t)i * HD;
  float s1 = 0.f, s2 = 0.f;
  for (int j = 0; j < HD; ++j) {
    float x = px[j] + ph[j];
    s1 += x;
    s2 += x * x;
  }
  float mu = s1 * (1.f / HD);
  float var = s2 * (1.f / HD) - mu * mu;
  float rstd = rsqrtf(var + LN_EPS);
  float dotv = 0.f;
  for (int j = 0; j < HD; ++j) {
    float x = px[j] + ph[j];
    dotv += (x - mu) * rstd * selW[j];
  }
  dotv += selb[0];
  out[mi[i]] = 1.f / (1.f + __expf(-dotv));
}

extern "C" void kernel_launch(void* const* d_in, const int* in_sizes, int n_in,
                              void* d_out, int out_size, void* d_ws, size_t ws_size,
                              hipStream_t stream) {
  const float* arrivals = (const float*)d_in[0];
  const float* departures = (const float*)d_in[1];
  const float* is_hard = (const float*)d_in[2];
  const int* mi = (const int*)d_in[3];
  const int* edge = (const int*)d_in[4];
  const int* tstep = (const int*)d_in[5];
  const float* W_emb1 = (const float*)d_in[6];
  const float* b_emb1 = (const float*)d_in[7];
  const float* W_emb2 = (const float*)d_in[8];
  const float* b_emb2 = (const float*)d_in[9];
  const float* lin_W = (const float*)d_in[10];
  const float* att_src = (const float*)d_in[11];
  const float* att_dst = (const float*)d_in[12];
  const float* gat_bias = (const float*)d_in[13];
  const float* selW = (const float*)d_in[14];
  const float* selb = (const float*)d_in[15];
  float* out = (float*)d_out;

  const int N = in_sizes[0];
  const int A = in_sizes[3];
  const int E = in_sizes[4] / 2;
  const int* esrc = edge;
  const int* edst = edge + E;

  // workspace carve-out (256B aligned slices)
  size_t off = 0;
  auto take = [&](size_t bytes) -> char* {
    char* p = (char*)d_ws + off;
    off += (bytes + 255) & ~(size_t)255;
    return p;
  };
  float* x0 = (float*)take((size_t)A * HD * 4);
  float* hbuf = (float*)take((size_t)A * HD * 4);
  float* hw = (float*)take((size_t)A * HD * 4);
  float* hacc = (float*)take((size_t)A * HD * 4);
  _Float16* h16 = (_Float16*)take((size_t)A * HD * 2);
  unsigned* wpack = (unsigned*)take(2048 * 4);
  float* a_s = (float*)take((size_t)A * 4);
  float* a_d = (float*)take((size_t)A * 4);
  unsigned* mseg = (unsigned*)take((size_t)A * 4);
  float* denom = (float*)take((size_t)A * 4);
  float* eraw = (float*)take((size_t)(E + A) * 4);
  float* Wf = (float*)take(2 * HD * 4);
  float* bf = (float*)take(HD * 4);
  (void)ws_size; (void)n_in; (void)out_size;

  const int tot = E + A;

  hipMemsetAsync(d_out, 0, (size_t)N * sizeof(float), stream);

  gat_fuse_emb<<<1, 64, 0, stream>>>(W_emb1, b_emb1, W_emb2, b_emb2, Wf, bf);
  gat_embed<<<(A + 127) / 128, 128, 0, stream>>>(arrivals, departures, is_hard, mi,
                                                 tstep, Wf, bf, x0, A);

  const float* h_in = x0;
  for (int l = 0; l < NLAYERS; ++l) {
    int nelem = A * HD;
    gat_f2h<<<(nelem / 4 + 255) / 256, 256, 0, stream>>>(h_in, h16, nelem);
    gat_pack_w<<<8, 256, 0, stream>>>(lin_W + (size_t)l * HD * HD, wpack);
    gat_gemm_wmma<<<(A + 31) / 32, 256, 0, stream>>>(h16, wpack, hw, A);
    gat_attn<<<(A + 127) / 128, 128, 0, stream>>>(hw, att_src + l * HD,
                                                  att_dst + l * HD, a_s, a_d, A);
    hipMemsetAsync(mseg, 0, (size_t)A * 4, stream);   // key 0 == -inf-most
    hipMemsetAsync(denom, 0, (size_t)A * 4, stream);
    hipMemsetAsync(hacc, 0, (size_t)A * HD * 4, stream);
    gat_edge_max<<<(tot + 255) / 256, 256, 0, stream>>>(esrc, edst, E, A, a_s, a_d,
                                                        eraw, mseg);
    gat_edge_exp<<<(tot + 255) / 256, 256, 0, stream>>>(edst, E, A, mseg, eraw, denom);
    gat_edge_aggr<<<(tot * HD + 255) / 256, 256, 0, stream>>>(esrc, edst, E, A, eraw,
                                                              denom, hw, hacc);
    gat_bias_act<<<(nelem + 255) / 256, 256, 0, stream>>>(hacc, gat_bias + l * HD,
                                                          hbuf, nelem, l < NLAYERS - 1);
    h_in = hbuf;
  }

  gat_final<<<(A + 127) / 128, 128, 0, stream>>>(x0, hbuf, mi, selW, selb, out, A);
}
